// AttnBlock_49246095016267
// MI455X (gfx1250) — compile-verified, matrix operand order checked by reading, not run on hardware
//
#include <hip/hip_runtime.h>
#include <hip/hip_bf16.h>

typedef __attribute__((ext_vector_type(16))) _Float16 v16h;
typedef __attribute__((ext_vector_type(8)))  _Float16 v8h;
typedef __attribute__((ext_vector_type(8)))  float    v8f;
typedef __attribute__((ext_vector_type(4)))  int      v4i;

#define C_DIM 512
#define N_DIM 4096
#define B_DIM 4
#define GROUPS 32
#define CPG 16
#define EPS 1e-6f

// CDNA5 async global->LDS copy path (guarded: falls back to plain copy)
#if defined(__HIP_DEVICE_COMPILE__) && \
    __has_builtin(__builtin_amdgcn_global_load_async_to_lds_b128) && \
    __has_builtin(__builtin_amdgcn_s_wait_asynccnt)
#define ASYNC_COPY 1
#else
#define ASYNC_COPY 0
#endif

#if ASYNC_COPY
typedef __attribute__((address_space(1))) v4i* as1_v4i_p;
typedef __attribute__((address_space(3))) v4i* as3_v4i_p;
static __device__ __forceinline__ void async_copy_b128(const void* g, void* l) {
  __builtin_amdgcn_global_load_async_to_lds_b128(
      (as1_v4i_p)(uintptr_t)g,
      (as3_v4i_p)(uintptr_t)(unsigned)(uintptr_t)l, 0, 0);
}
#endif

static __device__ __forceinline__ v8f wmma_f16(v16h a, v16h b, v8f c) {
  // D = A(16x32 f16) * B(32x16 f16) + C(16x16 f32)
  return __builtin_amdgcn_wmma_f32_16x16x32_f16(
      false, a, false, b, (short)0, c, false, false);
}

// ---------------------------------------------------------------------------
// Kernel 0: convert the 4 weight matrices (wq,wk,wv,wp) f32 -> f16
// ---------------------------------------------------------------------------
__global__ void cvt_weights(const float* __restrict__ wq, const float* __restrict__ wk,
                            const float* __restrict__ wv, const float* __restrict__ wp,
                            _Float16* __restrict__ w16) {
  size_t t = (size_t)blockIdx.x * blockDim.x + threadIdx.x;
  int m = (int)(t >> 18);
  int i = (int)(t & 262143);
  const float* src = (m == 0) ? wq : (m == 1) ? wk : (m == 2) ? wv : wp;
  w16[t] = (_Float16)src[i];
}

// ---------------------------------------------------------------------------
// Kernel 1: GroupNorm statistics. One block per (b,g).
// ---------------------------------------------------------------------------
__global__ void gn_stats(const float* __restrict__ x,
                         float* __restrict__ meanv, float* __restrict__ rstdv) {
  const int bg = blockIdx.x;
  const float* base = x + (size_t)bg * CPG * N_DIM;
  float s = 0.f, ss = 0.f;
  for (int i = threadIdx.x; i < CPG * N_DIM; i += 256) {
    float v = base[i];
    s += v; ss += v * v;
  }
  __shared__ float rs[256], rss[256];
  rs[threadIdx.x] = s; rss[threadIdx.x] = ss;
  __syncthreads();
  for (int off = 128; off > 0; off >>= 1) {
    if (threadIdx.x < off) {
      rs[threadIdx.x]  += rs[threadIdx.x + off];
      rss[threadIdx.x] += rss[threadIdx.x + off];
    }
    __syncthreads();
  }
  if (threadIdx.x == 0) {
    float inv_n = 1.0f / (float)(CPG * N_DIM);
    float mean = rs[0] * inv_n;
    float var  = rss[0] * inv_n - mean * mean;
    meanv[bg] = mean;
    rstdv[bg] = rsqrtf(var + EPS);
  }
}

// ---------------------------------------------------------------------------
// Kernel 2: apply GroupNorm, write H transposed [B][N][C] in f16
// ---------------------------------------------------------------------------
__global__ void gn_apply(const float* __restrict__ x,
                         const float* __restrict__ gns, const float* __restrict__ gnb,
                         const float* __restrict__ meanv, const float* __restrict__ rstdv,
                         _Float16* __restrict__ Ht) {
  size_t t = (size_t)blockIdx.x * blockDim.x + threadIdx.x;
  int c = (int)(t % C_DIM);
  size_t rem = t / C_DIM;
  int n = (int)(rem % N_DIM);
  int b = (int)(rem / N_DIM);
  int bg = b * GROUPS + (c >> 4);
  float v = x[((size_t)b * C_DIM + c) * N_DIM + n];
  float h = (v - meanv[bg]) * rstdv[bg] * gns[c] + gnb[c];
  Ht[((size_t)b * N_DIM + n) * C_DIM + c] = (_Float16)h;
}

// ---------------------------------------------------------------------------
// Kernel 3: QKV projection GEMM via WMMA; wave computes a 32x64 block
// (2 o-tiles x 4 n-tiles = 8 accs -> 1.5 b128 loads / WMMA).
//   which: 0 -> Q (scaled C^-0.5, transposed [N][C]); 1 -> K (transposed);
//          2 -> V ([C][N])
// ---------------------------------------------------------------------------
__global__ void qkv_gemm(const _Float16* __restrict__ w16,
                         const float* __restrict__ bq, const float* __restrict__ bk,
                         const float* __restrict__ bv,
                         const _Float16* __restrict__ Ht,
                         _Float16* __restrict__ Qt, _Float16* __restrict__ Kt,
                         _Float16* __restrict__ Vm) {
  const int b     = blockIdx.z;
  const int which = blockIdx.y;
  const int lane  = threadIdx.x & 31;
  const int wave  = threadIdx.x >> 5;
  const int tile  = blockIdx.x * 8 + wave;      // 0..1023
  const int op    = tile & 15;                  // o-pair (2 tiles of 16)
  const int n0    = (tile >> 4) * 64;           // pixel strip base

  const _Float16* W = w16 + (size_t)which * C_DIM * C_DIM;
  const float* bias = (which == 0) ? bq : (which == 1) ? bk : bv;
  const _Float16* H = Ht + (size_t)b * N_DIM * C_DIM;

  const int row = lane & 15;
  const int kh  = (lane >> 4) * 16;
  const int o0  = op * 32;

  v8f a00 = {}, a01 = {}, a02 = {}, a03 = {};
  v8f a10 = {}, a11 = {}, a12 = {}, a13 = {};
  for (int kc = 0; kc < C_DIM; kc += 32) {
    v16h w0 = *(const v16h*)(W + (size_t)(o0 +  0 + row) * C_DIM + kc + kh);
    v16h w1 = *(const v16h*)(W + (size_t)(o0 + 16 + row) * C_DIM + kc + kh);
    v16h h0 = *(const v16h*)(H + (size_t)(n0 +  0 + row) * C_DIM + kc + kh);
    v16h h1 = *(const v16h*)(H + (size_t)(n0 + 16 + row) * C_DIM + kc + kh);
    v16h h2 = *(const v16h*)(H + (size_t)(n0 + 32 + row) * C_DIM + kc + kh);
    v16h h3 = *(const v16h*)(H + (size_t)(n0 + 48 + row) * C_DIM + kc + kh);
    a00 = wmma_f16(w0, h0, a00);  a01 = wmma_f16(w0, h1, a01);
    a02 = wmma_f16(w0, h2, a02);  a03 = wmma_f16(w0, h3, a03);
    a10 = wmma_f16(w1, h0, a10);  a11 = wmma_f16(w1, h1, a11);
    a12 = wmma_f16(w1, h2, a12);  a13 = wmma_f16(w1, h3, a13);
  }

  v8f accs[2][4] = {{a00, a01, a02, a03}, {a10, a11, a12, a13}};
  #pragma unroll
  for (int u = 0; u < 2; ++u) {
    const int mb = o0 + u * 16 + (lane >> 4) * 8;
    if (which == 2) {
      _Float16* V = Vm + (size_t)b * C_DIM * N_DIM;
      #pragma unroll
      for (int t = 0; t < 4; ++t) {
        const int n = n0 + t * 16 + (lane & 15);
        #pragma unroll
        for (int r = 0; r < 8; ++r)
          V[(size_t)(mb + r) * N_DIM + n] = (_Float16)(accs[u][t][r] + bias[mb + r]);
      }
    } else {
      const float scale = (which == 0) ? 0.044194173824159216f : 1.0f;
      _Float16* D = ((which == 0) ? Qt : Kt) + (size_t)b * N_DIM * C_DIM;
      #pragma unroll
      for (int t = 0; t < 4; ++t) {
        const int n = n0 + t * 16 + (lane & 15);
        v8h o;
        #pragma unroll
        for (int r = 0; r < 8; ++r)
          o[r] = (_Float16)((accs[u][t][r] + bias[mb + r]) * scale);
        *(v8h*)(D + (size_t)n * C_DIM + mb) = o;
      }
    }
  }
}

// ---------------------------------------------------------------------------
// Kernel 4: fused attention for one 32-row query block (2 i-tiles).
//  K tiles feed both i-tiles in phase 1; V tiles feed both in phase 3.
//  LDS: sQ 32KB + sP 256KB + reductions ~1.3KB (~290KB of the 320KB WGP LDS).
// ---------------------------------------------------------------------------
__global__ void attn_fused(const _Float16* __restrict__ Qt,
                           const _Float16* __restrict__ Kt,
                           const _Float16* __restrict__ Vm,
                           _Float16* __restrict__ Ot) {
  const int b  = blockIdx.y;
  const int i0 = blockIdx.x * 32;
  extern __shared__ char smem[];
  _Float16* sQ  = (_Float16*)smem;                            // [32][512]  32KB
  _Float16* sP  = (_Float16*)(smem + 32 * C_DIM * 2);         // [32][4096] 256KB
  float* sRed   = (float*)(smem + 32 * C_DIM * 2 + 32 * N_DIM * 2);  // [32][8]
  float* sMax   = sRed + 256;                                 // [32]
  float* sSum   = sMax + 32;                                  // [32]

  const int tid  = threadIdx.x;
  const int lane = tid & 31;
  const int wave = tid >> 5;
  const int row  = lane & 15;
  const int kh   = (lane >> 4) * 16;

  const _Float16* Qb = Qt + (size_t)b * N_DIM * C_DIM;
  const _Float16* Kb = Kt + (size_t)b * N_DIM * C_DIM;
  const _Float16* Vb = Vm + (size_t)b * C_DIM * N_DIM;

  // ---- Stage Q tile (32x512 f16 = 32KB) into LDS ----
  for (int t = tid; t < 2048; t += 256) {
    const int r  = t >> 6;             // row 0..31
    const int c8 = (t & 63) * 8;       // 8 f16 = 16B per chunk
    const _Float16* g = Qb + (size_t)(i0 + r) * C_DIM + c8;
    _Float16* l = sQ + r * C_DIM + c8;
#if ASYNC_COPY
    async_copy_b128(g, l);
#else
    *(v8h*)l = *(const v8h*)g;
#endif
  }
#if ASYNC_COPY
  __builtin_amdgcn_s_wait_asynccnt(0);
#endif
  __syncthreads();

  // ---- Phase 1: S = Q^T K; 2 i-tiles x 4 j-tiles per wave iteration ----
  for (int jq = wave; jq < N_DIM / 64; jq += 8) {
    const int jt = jq * 64;
    v8f s00 = {}, s01 = {}, s02 = {}, s03 = {};
    v8f s10 = {}, s11 = {}, s12 = {}, s13 = {};
    for (int kc = 0; kc < C_DIM; kc += 32) {
      v16h q0 = *(const v16h*)(sQ + (size_t)( 0 + row) * C_DIM + kc + kh);
      v16h q1 = *(const v16h*)(sQ + (size_t)(16 + row) * C_DIM + kc + kh);
      v16h k0 = *(const v16h*)(Kb + (size_t)(jt +  0 + row) * C_DIM + kc + kh);
      v16h k1 = *(const v16h*)(Kb + (size_t)(jt + 16 + row) * C_DIM + kc + kh);
      v16h k2 = *(const v16h*)(Kb + (size_t)(jt + 32 + row) * C_DIM + kc + kh);
      v16h k3 = *(const v16h*)(Kb + (size_t)(jt + 48 + row) * C_DIM + kc + kh);
      s00 = wmma_f16(q0, k0, s00);  s01 = wmma_f16(q0, k1, s01);
      s02 = wmma_f16(q0, k2, s02);  s03 = wmma_f16(q0, k3, s03);
      s10 = wmma_f16(q1, k0, s10);  s11 = wmma_f16(q1, k1, s11);
      s12 = wmma_f16(q1, k2, s12);  s13 = wmma_f16(q1, k3, s13);
    }
    const int n  = lane & 15;
    const int mb = (lane >> 4) * 8;
    v8f accs[2][4] = {{s00, s01, s02, s03}, {s10, s11, s12, s13}};
    #pragma unroll
    for (int u = 0; u < 2; ++u)
      #pragma unroll
      for (int t = 0; t < 4; ++t)
        #pragma unroll
        for (int r = 0; r < 8; ++r)
          sP[(size_t)(u * 16 + mb + r) * N_DIM + jt + t * 16 + n] =
              (_Float16)accs[u][t][r];
  }
  __syncthreads();

  // ---- Phase 2: softmax over 32 rows (8 threads per row) ----
  const int srow = tid >> 3;
  const int scol = tid & 7;
  float mx = -1e30f;
  for (int j = scol; j < N_DIM; j += 8)
    mx = fmaxf(mx, (float)sP[(size_t)srow * N_DIM + j]);
  sRed[srow * 8 + scol] = mx;
  __syncthreads();
  if (scol == 0) {
    float m = sRed[srow * 8];
    for (int t = 1; t < 8; ++t) m = fmaxf(m, sRed[srow * 8 + t]);
    sMax[srow] = m;
  }
  __syncthreads();
  const float rm = sMax[srow];
  float s = 0.f;
  for (int j = scol; j < N_DIM; j += 8) {
    float e = __expf((float)sP[(size_t)srow * N_DIM + j] - rm);
    sP[(size_t)srow * N_DIM + j] = (_Float16)e;
    s += e;
  }
  sRed[srow * 8 + scol] = s;
  __syncthreads();
  if (scol == 0) {
    float t = 0.f;
    for (int q = 0; q < 8; ++q) t += sRed[srow * 8 + q];
    sSum[srow] = t;
  }
  __syncthreads();

  // ---- Phase 3: O^T(c,i) += V(c,j) * P^T(j,i); wave owns 64 channels,
  //      V tiles shared across the 2 i-tiles ----
  const int c0 = wave * 64;
  v8f o00 = {}, o01 = {}, o02 = {}, o03 = {};
  v8f o10 = {}, o11 = {}, o12 = {}, o13 = {};
  for (int j = 0; j < N_DIM; j += 32) {
    v16h p0 = *(const v16h*)(sP + (size_t)( 0 + (lane & 15)) * N_DIM + j + kh);
    v16h p1 = *(const v16h*)(sP + (size_t)(16 + (lane & 15)) * N_DIM + j + kh);
    v16h a0 = *(const v16h*)(Vb + (size_t)(c0 +  0 + row) * N_DIM + j + kh);
    v16h a1 = *(const v16h*)(Vb + (size_t)(c0 + 16 + row) * N_DIM + j + kh);
    v16h a2 = *(const v16h*)(Vb + (size_t)(c0 + 32 + row) * N_DIM + j + kh);
    v16h a3 = *(const v16h*)(Vb + (size_t)(c0 + 48 + row) * N_DIM + j + kh);
    o00 = wmma_f16(a0, p0, o00);  o10 = wmma_f16(a0, p1, o10);
    o01 = wmma_f16(a1, p0, o01);  o11 = wmma_f16(a1, p1, o11);
    o02 = wmma_f16(a2, p0, o02);  o12 = wmma_f16(a2, p1, o12);
    o03 = wmma_f16(a3, p0, o03);  o13 = wmma_f16(a3, p1, o13);
  }

  const int mb2 = (lane >> 4) * 8;
  _Float16* Ob = Ot + (size_t)b * N_DIM * C_DIM;
  v8f accs[2][4] = {{o00, o01, o02, o03}, {o10, o11, o12, o13}};
  #pragma unroll
  for (int u = 0; u < 2; ++u) {
    const int n_i = i0 + u * 16 + (lane & 15);
    const float inv = 1.0f / sSum[u * 16 + (lane & 15)];
    #pragma unroll
    for (int t = 0; t < 4; ++t) {
      v8h o;
      #pragma unroll
      for (int r = 0; r < 8; ++r)
        o[r] = (_Float16)(accs[u][t][r] * inv);
      *(v8h*)(Ob + (size_t)n_i * C_DIM + c0 + t * 16 + mb2) = o;
    }
  }
}

// ---------------------------------------------------------------------------
// Kernel 5: projection GEMM + bias + residual (32x64 block per wave)
// ---------------------------------------------------------------------------
__global__ void proj_gemm(const _Float16* __restrict__ w16,
                          const float* __restrict__ bp,
                          const _Float16* __restrict__ Ot,
                          const float* __restrict__ x,
                          float* __restrict__ out) {
  const int b    = blockIdx.z;
  const int lane = threadIdx.x & 31;
  const int wave = threadIdx.x >> 5;
  const int tile = blockIdx.x * 8 + wave;       // 0..1023
  const int op   = tile & 15;
  const int n0   = (tile >> 4) * 64;

  const _Float16* W = w16 + (size_t)3 * C_DIM * C_DIM;
  const _Float16* O = Ot + (size_t)b * N_DIM * C_DIM;

  const int row = lane & 15;
  const int kh  = (lane >> 4) * 16;
  const int o0  = op * 32;

  v8f a00 = {}, a01 = {}, a02 = {}, a03 = {};
  v8f a10 = {}, a11 = {}, a12 = {}, a13 = {};
  for (int kc = 0; kc < C_DIM; kc += 32) {
    v16h w0 = *(const v16h*)(W + (size_t)(o0 +  0 + row) * C_DIM + kc + kh);
    v16h w1 = *(const v16h*)(W + (size_t)(o0 + 16 + row) * C_DIM + kc + kh);
    v16h h0 = *(const v16h*)(O + (size_t)(n0 +  0 + row) * C_DIM + kc + kh);
    v16h h1 = *(const v16h*)(O + (size_t)(n0 + 16 + row) * C_DIM + kc + kh);
    v16h h2 = *(const v16h*)(O + (size_t)(n0 + 32 + row) * C_DIM + kc + kh);
    v16h h3 = *(const v16h*)(O + (size_t)(n0 + 48 + row) * C_DIM + kc + kh);
    a00 = wmma_f16(w0, h0, a00);  a01 = wmma_f16(w0, h1, a01);
    a02 = wmma_f16(w0, h2, a02);  a03 = wmma_f16(w0, h3, a03);
    a10 = wmma_f16(w1, h0, a10);  a11 = wmma_f16(w1, h1, a11);
    a12 = wmma_f16(w1, h2, a12);  a13 = wmma_f16(w1, h3, a13);
  }

  v8f accs[2][4] = {{a00, a01, a02, a03}, {a10, a11, a12, a13}};
  #pragma unroll
  for (int u = 0; u < 2; ++u) {
    const int mb = o0 + u * 16 + (lane >> 4) * 8;
    #pragma unroll
    for (int t = 0; t < 4; ++t) {
      const int n = n0 + t * 16 + (lane & 15);
      #pragma unroll
      for (int r = 0; r < 8; ++r) {
        const int o = mb + r;
        const size_t idx = ((size_t)b * C_DIM + o) * N_DIM + n;
        out[idx] = accs[u][t][r] + bp[o] + x[idx];
      }
    }
  }
}

// ---------------------------------------------------------------------------
// Host launcher
// ---------------------------------------------------------------------------
extern "C" void kernel_launch(void* const* d_in, const int* in_sizes, int n_in,
                              void* d_out, int out_size, void* d_ws, size_t ws_size,
                              hipStream_t stream) {
  const float* x    = (const float*)d_in[0];
  const float* gns  = (const float*)d_in[1];
  const float* gnb  = (const float*)d_in[2];
  const float* wq   = (const float*)d_in[3];
  const float* bq   = (const float*)d_in[4];
  const float* wk   = (const float*)d_in[5];
  const float* bk   = (const float*)d_in[6];
  const float* wv   = (const float*)d_in[7];
  const float* bv   = (const float*)d_in[8];
  const float* wp   = (const float*)d_in[9];
  const float* bp   = (const float*)d_in[10];
  float* out = (float*)d_out;

  char* p = (char*)d_ws;
  _Float16* w16 = (_Float16*)p;  p += (size_t)4 * C_DIM * C_DIM * sizeof(_Float16);
  float* meanv  = (float*)p;     p += 512;
  float* rstdv  = (float*)p;     p += 512;
  const size_t MAT = (size_t)B_DIM * N_DIM * C_DIM * sizeof(_Float16);
  _Float16* Ht = (_Float16*)p;   p += MAT;
  _Float16* Qt = (_Float16*)p;   p += MAT;
  _Float16* Kt = (_Float16*)p;   p += MAT;
  _Float16* Vm = (_Float16*)p;   p += MAT;
  _Float16* Ot = (_Float16*)p;   p += MAT;

  cvt_weights<<<(4 * C_DIM * C_DIM) / 256, 256, 0, stream>>>(wq, wk, wv, wp, w16);
  gn_stats<<<B_DIM * GROUPS, 256, 0, stream>>>(x, meanv, rstdv);
  {
    size_t total = (size_t)B_DIM * C_DIM * N_DIM;
    gn_apply<<<(unsigned)(total / 256), 256, 0, stream>>>(x, gns, gnb, meanv, rstdv, Ht);
  }
  {
    dim3 grid(128, 3, B_DIM);     // 1024 blocks of 32x64 per (mat, batch)
    qkv_gemm<<<grid, 256, 0, stream>>>(w16, bq, bk, bv, Ht, Qt, Kt, Vm);
  }
  {
    dim3 grid(N_DIM / 32, B_DIM);
    size_t shmem = (size_t)32 * C_DIM * sizeof(_Float16)    // Q tile   32KB
                 + (size_t)32 * N_DIM * sizeof(_Float16)    // scores  256KB
                 + (256 + 32 + 32) * sizeof(float);         // reductions
    attn_fused<<<grid, 256, shmem, stream>>>(Qt, Kt, Vm, Ot);
  }
  {
    dim3 grid(128, 1, B_DIM);
    proj_gemm<<<grid, 256, 0, stream>>>(w16, bp, Ot, x, out);
  }
}
